// MultiAttn_75960791597072
// MI455X (gfx1250) — compile-verified, matrix operand order checked by reading, not run on hardware
//
#include <hip/hip_runtime.h>

typedef __attribute__((ext_vector_type(16))) _Float16 v16h;
typedef __attribute__((ext_vector_type(8)))  _Float16 v8h;
typedef __attribute__((ext_vector_type(8)))  float    v8f;

static constexpr int B_ = 2, S_ = 2048, D_ = 1024, H_ = 16, HS = 64;

#define WMMA_F16(a, b, c) \
  __builtin_amdgcn_wmma_f32_16x16x32_f16(false, (a), false, (b), (short)0, (c), false, false)

// ---------------------------------------------------------------------------
// Fragment loader (ISA 7.12.2 layouts, wave32)
// A-matrix 16x32 f16: lane L holds row M=L%16; element e -> K = 8*(L/16) + (e<8 ? e : e+8)
// B-matrix 32x16 f16: lane L holds col N=L%16; same per-element K mapping.
// C/D 16x16 f32: lane L holds col N=L%16; element e -> M = e + 8*(L/16).
// Both A and B fragments reduce to two contiguous 16-byte loads per lane.
// ---------------------------------------------------------------------------
__device__ __forceinline__ v16h load_frag_row(const _Float16* p, int stride,
                                              int row, int k0, int lane) {
  const _Float16* q = p + (size_t)row * stride + k0 + ((lane >> 4) << 3);
  v8h lo = *(const v8h*)(q);        // K = base .. base+7
  v8h hi = *(const v8h*)(q + 16);   // K = base+16 .. base+23
  v16h r;
#pragma unroll
  for (int i = 0; i < 8; ++i) { r[i] = lo[i]; r[i + 8] = hi[i]; }
  return r;
}

// ---------------------------------------------------------------------------
// K0: f32 -> f16 staging
// ---------------------------------------------------------------------------
__global__ void cvt_f16_kernel(const float* __restrict__ s, _Float16* __restrict__ d, int n) {
  int i = blockIdx.x * blockDim.x + threadIdx.x;
  int stride = gridDim.x * blockDim.x;
  for (; i < n; i += stride) d[i] = (_Float16)s[i];
}

// ---------------------------------------------------------------------------
// K1: QKV projection (4096x3072x1024). Wave tile = 16x64 (one head/part block).
// Epilogue: pre-RoPE K,V f32 cache + head-indexed RoPE -> qh, kh (key-major)
// and vT (dim-major, hs x S per head) f16 staging.
// ---------------------------------------------------------------------------
__global__ void qkv_kernel(const _Float16* __restrict__ xh, const _Float16* __restrict__ wph,
                           const float* __restrict__ freqs, float* __restrict__ out_kv,
                           _Float16* __restrict__ qh, _Float16* __restrict__ kh,
                           _Float16* __restrict__ vT) {
  const int lane = threadIdx.x & 31;
  const int wave = threadIdx.x >> 5;
  const int NT = (3 * D_) / 64;  // 48 N-tiles of width 64
  int tile = blockIdx.x * 4 + wave;
  int m0 = (tile / NT) * 16;
  int n0 = (tile % NT) * 64;

  v8f c[4] = {};
#pragma unroll 4
  for (int k0 = 0; k0 < D_; k0 += 32) {
    v16h a = load_frag_row(xh, D_, m0 + (lane & 15), k0, lane);
#pragma unroll
    for (int t = 0; t < 4; ++t) {
      v16h b = load_frag_row(wph, D_, n0 + 16 * t + (lane & 15), k0, lane);
      c[t] = WMMA_F16(a, b, c[t]);
    }
  }

  // 64-wide tile == exactly one (head, part) block: 3072 = H * [q(64)|k(64)|v(64)]
  int blk  = n0 / 64;
  int h    = blk / 3;
  int part = blk % 3;        // 0=q, 1=k, 2=v
  int n    = lane & 15;
  int half = lane >> 4;

#pragma unroll
  for (int t = 0; t < 4; ++t) {
    int d = 16 * t + n;
    // RoPE quirk: angle row indexed by HEAD (freqs[:H]), not position.
    int j = d >> 1;
    float cs = freqs[(h * 32 + j) * 2 + 0];
    float sn = freqs[(h * 32 + j) * 2 + 1];
#pragma unroll
    for (int e = 0; e < 8; ++e) {
      int m  = m0 + e + 8 * half;       // global row = b*S + s
      int b_ = m / S_;
      int s_ = m % S_;
      float v  = c[t][e];
      float pv = __shfl_xor(v, 1, 32);  // pair partner (d ^ 1)
      float roped = v * cs + ((d & 1) ? pv * sn : -pv * sn);
      int bh = b_ * H_ + h;
      if (part == 0) {
        qh[((size_t)bh * S_ + s_) * HS + d] = (_Float16)roped;
      } else if (part == 1) {
        out_kv[(((0 * B_ + b_) * H_ + h) * S_ + s_) * HS + d] = v;  // pre-RoPE cache
        kh[((size_t)bh * S_ + s_) * HS + d] = (_Float16)roped;
      } else {
        out_kv[(((1 * B_ + b_) * H_ + h) * S_ + s_) * HS + d] = v;
        vT[((size_t)bh * HS + d) * S_ + s_] = (_Float16)v;          // transposed for P.V
      }
    }
  }
}

// ---------------------------------------------------------------------------
// K2: flash attention. One wave per 16-query tile; stream 32-key blocks.
// ---------------------------------------------------------------------------
__global__ void attn_kernel(const _Float16* __restrict__ qh, const _Float16* __restrict__ kh,
                            const _Float16* __restrict__ vT, _Float16* __restrict__ oh) {
  __shared__ __align__(16) _Float16 plds_all[4][16 * 32];
  const int lane = threadIdx.x & 31;
  const int wave = threadIdx.x >> 5;
  _Float16* plds = &plds_all[wave][0];

  int tile = blockIdx.x * 4 + wave;   // 0..4095
  int qt = tile & 127;                // S/16 query tiles
  int bh = tile >> 7;                 // (b*H + h)
  int b_ = bh / H_, h = bh % H_;
  const _Float16* Qp = qh + (size_t)bh * S_ * HS;
  const _Float16* Kp = kh + (size_t)bh * S_ * HS;
  const _Float16* Vt = vT + (size_t)bh * HS * S_;

  int q0 = qt * 16;
  int n = lane & 15, half = lane >> 4;

  v16h aq0 = load_frag_row(Qp, HS, q0 + n, 0, lane);
  v16h aq1 = load_frag_row(Qp, HS, q0 + n, 32, lane);

  float mrow[8], lrow[8];
  v8f acc[4] = {};
#pragma unroll
  for (int e = 0; e < 8; ++e) { mrow[e] = -__builtin_inff(); lrow[e] = 0.f; }

  for (int k0 = 0; k0 <= q0 + 15; k0 += 32) {
    // scores: Q(16x64) . K^T for 32 keys -> two 16x16 C tiles
    v8f s0 = {}, s1 = {};
    {
      v16h bk = load_frag_row(Kp, HS, k0 + n, 0, lane);
      s0 = WMMA_F16(aq0, bk, s0);
      bk = load_frag_row(Kp, HS, k0 + n, 32, lane);
      s0 = WMMA_F16(aq1, bk, s0);
      bk = load_frag_row(Kp, HS, k0 + 16 + n, 0, lane);
      s1 = WMMA_F16(aq0, bk, s1);
      bk = load_frag_row(Kp, HS, k0 + 16 + n, 32, lane);
      s1 = WMMA_F16(aq1, bk, s1);
    }
#pragma unroll
    for (int e = 0; e < 8; ++e) {
      int qi = q0 + e + 8 * half;
      float v0 = (k0 + n      > qi) ? -65504.f : s0[e] * 0.125f;  // 1/sqrt(64), MIN_FP16 mask
      float v1 = (k0 + 16 + n > qi) ? -65504.f : s1[e] * 0.125f;
      float tm = fmaxf(v0, v1);
#pragma unroll
      for (int msk = 8; msk >= 1; msk >>= 1) tm = fmaxf(tm, __shfl_xor(tm, msk, 32));
      float mn = fmaxf(mrow[e], tm);
      float sc = __expf(mrow[e] - mn);
      float p0 = __expf(v0 - mn);
      float p1 = __expf(v1 - mn);
      float rs = p0 + p1;
#pragma unroll
      for (int msk = 8; msk >= 1; msk >>= 1) rs += __shfl_xor(rs, msk, 32);
      lrow[e] = lrow[e] * sc + rs;
      mrow[e] = mn;
#pragma unroll
      for (int t = 0; t < 4; ++t) acc[t][e] = acc[t][e] * sc;
      int r = e + 8 * half;
      plds[r * 32 + n]      = (_Float16)p0;   // C-layout -> row-major P in LDS
      plds[r * 32 + n + 16] = (_Float16)p1;
    }
    asm volatile("s_wait_dscnt 0x0" ::: "memory");   // cross-lane LDS RAW fence (same wave)
    v16h pa = load_frag_row(plds, 32, n, 0, lane);   // P as A-fragment (16x32)
#pragma unroll
    for (int t = 0; t < 4; ++t) {
      // V^T is dim-major: B[k=key][n=dim] = two contiguous b128 loads along keys
      v16h bv = load_frag_row(Vt, S_, t * 16 + n, k0, lane);
      acc[t] = WMMA_F16(pa, bv, acc[t]);
    }
    // note: no trailing fence needed -- the WMMA consuming `pa` already forces
    // the DS-load wait, and same-wave LDS ops stay in order thereafter.
  }

#pragma unroll
  for (int e = 0; e < 8; ++e) {
    float inv = 1.0f / lrow[e];
    int s_ = q0 + e + 8 * half;
#pragma unroll
    for (int t = 0; t < 4; ++t) {
      int d = t * 16 + n;
      oh[((size_t)(b_ * S_ + s_)) * D_ + h * HS + d] = (_Float16)(acc[t][e] * inv);
    }
  }
}

// ---------------------------------------------------------------------------
// K3: output projection o @ W_ff^T (4096 x 1024 x 1024) -> f32. 16x64 wave tile.
// ---------------------------------------------------------------------------
__global__ void ffn_kernel(const _Float16* __restrict__ oh, const _Float16* __restrict__ wfh,
                           float* __restrict__ out) {
  const int lane = threadIdx.x & 31;
  const int wave = threadIdx.x >> 5;
  const int NT = D_ / 64;  // 16 N-tiles of width 64
  int tile = blockIdx.x * 4 + wave;
  int m0 = (tile / NT) * 16;
  int n0 = (tile % NT) * 64;

  v8f c[4] = {};
#pragma unroll 4
  for (int k0 = 0; k0 < D_; k0 += 32) {
    v16h a = load_frag_row(oh, D_, m0 + (lane & 15), k0, lane);
#pragma unroll
    for (int t = 0; t < 4; ++t) {
      v16h b = load_frag_row(wfh, D_, n0 + 16 * t + (lane & 15), k0, lane);
      c[t] = WMMA_F16(a, b, c[t]);
    }
  }
  int n = lane & 15, half = lane >> 4;
#pragma unroll
  for (int e = 0; e < 8; ++e) {
    int m = m0 + e + 8 * half;
#pragma unroll
    for (int t = 0; t < 4; ++t) out[(size_t)m * D_ + n0 + 16 * t + n] = c[t][e];
  }
}

// ---------------------------------------------------------------------------
extern "C" void kernel_launch(void* const* d_in, const int* in_sizes, int n_in,
                              void* d_out, int out_size, void* d_ws, size_t ws_size,
                              hipStream_t stream) {
  const float* x     = (const float*)d_in[0];
  const float* freqs = (const float*)d_in[1];
  // d_in[2] = mask (causal, reconstructed analytically), d_in[3] = num_head (fixed 16)
  const float* Wp    = (const float*)d_in[4];
  const float* Wf    = (const float*)d_in[5];

  float* out_ff = (float*)d_out;                          // (B,S,D) f32
  float* out_kv = out_ff + (size_t)B_ * S_ * D_;          // (2,B,H,S,hs) f32

  char* ws = (char*)d_ws;
  _Float16* xh  = (_Float16*)ws; ws += (size_t)B_ * S_ * D_ * 2;
  _Float16* wph = (_Float16*)ws; ws += (size_t)3 * D_ * D_ * 2;
  _Float16* wfh = (_Float16*)ws; ws += (size_t)D_ * D_ * 2;
  _Float16* qh  = (_Float16*)ws; ws += (size_t)B_ * H_ * S_ * HS * 2;
  _Float16* kh  = (_Float16*)ws; ws += (size_t)B_ * H_ * S_ * HS * 2;
  _Float16* vT  = (_Float16*)ws; ws += (size_t)B_ * H_ * S_ * HS * 2;
  _Float16* oh  = (_Float16*)ws; ws += (size_t)B_ * S_ * D_ * 2;

  int nx = B_ * S_ * D_, np = 3 * D_ * D_, nf = D_ * D_;
  cvt_f16_kernel<<<dim3((nx + 255) / 256), dim3(256), 0, stream>>>(x,  xh,  nx);
  cvt_f16_kernel<<<dim3((np + 255) / 256), dim3(256), 0, stream>>>(Wp, wph, np);
  cvt_f16_kernel<<<dim3((nf + 255) / 256), dim3(256), 0, stream>>>(Wf, wfh, nf);

  // QKV: 256 M-tiles x 48 N-tiles(64-wide) = 12288 wave-tiles, 4 waves/block
  qkv_kernel<<<dim3(12288 / 4), dim3(128), 0, stream>>>(xh, wph, freqs, out_kv, qh, kh, vT);

  // Attention: B*H*(S/16) = 4096 wave-tiles
  attn_kernel<<<dim3(4096 / 4), dim3(128), 0, stream>>>(qh, kh, vT, oh);

  // Output projection: 256 M-tiles x 16 N-tiles(64-wide) = 4096 wave-tiles
  ffn_kernel<<<dim3(4096 / 4), dim3(128), 0, stream>>>(oh, wfh, out_ff);
}
